// MultiHeadCrossAttention_58454504898882
// MI455X (gfx1250) — matrix operand order repaired and checked by reading, not compiled, not run on hardware
//
#include <hip/hip_runtime.h>
#include <hip/hip_bf16.h>

typedef __bf16 bf16;
typedef __attribute__((ext_vector_type(16))) __bf16 v16bf;
typedef __attribute__((ext_vector_type(8)))  float          v8f;
typedef __attribute__((ext_vector_type(8)))  unsigned short v8u;

#define T_DIM 1024
#define EMBD  1024
#define NH    16
#define HEADD 64
#define BATCH 4

// ---------------------------------------------------------------------------
// CDNA5 async global->LDS copy (ASYNCcnt) helpers.
// ---------------------------------------------------------------------------
static __device__ __forceinline__ void async_copy_b128(unsigned lds_off, const bf16* g) {
  // the vdst VGPR holds the LDS address; low 32 bits of a generic __shared__ pointer
  // are the LDS offset within the workgroup allocation.
  asm volatile("global_load_async_to_lds_b128 %0, %1, off"
               :: "v"(lds_off), "v"(g) : "memory");
}

template <int N>
static __device__ __forceinline__ void wait_async() {
#if __has_builtin(__builtin_amdgcn_s_wait_asynccnt)
  __builtin_amdgcn_s_wait_asynccnt((unsigned short)N);
#else
  asm volatile("s_wait_asynccnt %0" :: "i"(N) : "memory");
#endif
}

// ---------------------------------------------------------------------------
// WMMA fragment layouts (ISA 7.12.2), bf16 16x16x32.
// A (16x32): lane L<16 -> row M=L, K = {0..7, 16..23}; L>=16 -> K = {8..15, 24..31}
// B (32x16): lane L<16 -> col N=L, K = {0..7, 16..23}; L>=16 -> col N=L-16, K = {8..15, 24..31}
// ---------------------------------------------------------------------------
static __device__ __forceinline__ v16bf frag_a(const bf16* __restrict__ p0, int ld,
                                               int m0, int k0, int lane) {
  int r  = lane & 15;
  int ko = (lane >> 4) << 3;  // 0 or 8
  const bf16* p = p0 + (size_t)(m0 + r) * ld + (k0 + ko);
  union { struct { v8u lo, hi; } h; v16bf v; } u;
  u.h.lo = *(const v8u*)p;
  u.h.hi = *(const v8u*)(p + 16);
  return u.v;
}

static __device__ __forceinline__ v16bf frag_b(const bf16* __restrict__ p0, int ld,
                                               int n0, int k0, int lane) {
  int c  = lane & 15;
  int ko = (lane >> 4) << 3;  // 0 or 8
  const bf16* p = p0 + (size_t)(n0 + c) * ld + (k0 + ko);
  union { struct { v8u lo, hi; } h; v16bf v; } u;
  u.h.lo = *(const v8u*)p;
  u.h.hi = *(const v8u*)(p + 16);
  return u.v;
}

static __device__ __forceinline__ void zero_acc(v8f acc[4][4]) {
  v8f z = {0.f, 0.f, 0.f, 0.f, 0.f, 0.f, 0.f, 0.f};
#pragma unroll
  for (int i = 0; i < 4; ++i)
#pragma unroll
    for (int j = 0; j < 4; ++j) acc[i][j] = z;
}

// ---------------------------------------------------------------------------
// Direct-from-global 64x64 wave GEMM (used for K=64 score GEMM only).
// ---------------------------------------------------------------------------
static __device__ __forceinline__ void wave_gemm_64x64(
    const bf16* __restrict__ A, int lda, const bf16* __restrict__ B, int ldb,
    int m0, int n0, int K, int lane, v8f acc[4][4]) {
  for (int k0 = 0; k0 < K; k0 += 32) {
    v16bf a[4], b[4];
#pragma unroll
    for (int i = 0; i < 4; ++i) a[i] = frag_a(A, lda, m0 + 16 * i, k0, lane);
#pragma unroll
    for (int j = 0; j < 4; ++j) b[j] = frag_b(B, ldb, n0 + 16 * j, k0, lane);
#pragma unroll
    for (int i = 0; i < 4; ++i)
#pragma unroll
      for (int j = 0; j < 4; ++j)
        acc[i][j] = __builtin_amdgcn_wmma_f32_16x16x32_bf16(
            false, a[i], false, b[j], (short)0, acc[i][j], false, false);
  }
}

// ---------------------------------------------------------------------------
// Async-LDS double-buffered block GEMM.
// Block = WM x WN wave32s (wave tile 64x64), block tile (WM*64) x (WN*64),
// K-step 32. LDS rows padded to 40 bf16 (80 B) -> conflict-free ds_load_b128.
// Pipeline per K-step: issue stage k+1 (ASYNCcnt), s_wait_asynccnt, barrier,
// 16 WMMAs from LDS, barrier.
// ---------------------------------------------------------------------------
template <int WM, int WN>
static __device__ __forceinline__ void block_gemm_async(
    const bf16* __restrict__ A, int lda, const bf16* __restrict__ B, int ldb,
    int gm0, int gn0, int K, v8f acc[4][4]) {
  constexpr int BM = WM * 64;
  constexpr int BN = WN * 64;
  constexpr int LDS_LD = 40;  // 32 + 8 pad (16 B)
  constexpr int NTHR = WM * WN * 32;
  constexpr int OPS = (BM * 4 + BN * 4) / NTHR;  // async ops per thread per stage

  __shared__ __align__(16) bf16 sA[2][BM * LDS_LD];
  __shared__ __align__(16) bf16 sB[2][BN * LDS_LD];

  const int tid = threadIdx.x;
  const int lane = tid & 31;
  const int wave = tid >> 5;
  const int m0 = (wave / WN) * 64;
  const int n0 = (wave % WN) * 64;

  auto issue = [&](int buf, int k0) {
#pragma unroll
    for (int ch = 0; ch < BM * 4 / NTHR; ++ch) {
      int c = ch * NTHR + tid;
      int row = c >> 2, col = (c & 3) << 3;
      async_copy_b128((unsigned)(uintptr_t)&sA[buf][row * LDS_LD + col],
                      A + (size_t)(gm0 + row) * lda + k0 + col);
    }
#pragma unroll
    for (int ch = 0; ch < BN * 4 / NTHR; ++ch) {
      int c = ch * NTHR + tid;
      int row = c >> 2, col = (c & 3) << 3;
      async_copy_b128((unsigned)(uintptr_t)&sB[buf][row * LDS_LD + col],
                      B + (size_t)(gn0 + row) * ldb + k0 + col);
    }
  };

  const int nk = K >> 5;
  issue(0, 0);
  for (int k = 0; k < nk; ++k) {
    if (k + 1 < nk) {
      issue((k + 1) & 1, (k + 1) << 5);
      wait_async<OPS>();  // stage k arrived; stage k+1 still in flight
    } else {
      wait_async<0>();
    }
    __syncthreads();  // all waves' stage k visible in LDS
    const bf16* cA = sA[k & 1];
    const bf16* cB = sB[k & 1];
    v16bf a[4], b[4];
#pragma unroll
    for (int i = 0; i < 4; ++i) a[i] = frag_a(cA, LDS_LD, m0 + 16 * i, 0, lane);
#pragma unroll
    for (int j = 0; j < 4; ++j) b[j] = frag_b(cB, LDS_LD, n0 + 16 * j, 0, lane);
#pragma unroll
    for (int i = 0; i < 4; ++i)
#pragma unroll
      for (int j = 0; j < 4; ++j)
        acc[i][j] = __builtin_amdgcn_wmma_f32_16x16x32_bf16(
            false, a[i], false, b[j], (short)0, acc[i][j], false, false);
    __syncthreads();  // buffer free for stage k+2
  }
}

// ---------------------------------------------------------------------------
// f32 -> bf16 conversion
// ---------------------------------------------------------------------------
__global__ void cvt_f32_bf16(const float* __restrict__ src, bf16* __restrict__ dst, int n) {
  int i = blockIdx.x * blockDim.x + threadIdx.x;
  int stride = gridDim.x * blockDim.x;
  for (; i < n; i += stride) dst[i] = (bf16)src[i];
}

// ---------------------------------------------------------------------------
// Projection GEMM: C = A[M x K] * W[N x K]^T (y = x W^T), bf16 out.
// mode 0: out[m*N + n] = val * scale  (Q/K)
// mode 1: V written transposed per head: Vt[b][h][d][s]
// ---------------------------------------------------------------------------
__global__ __launch_bounds__(128) void gemm_proj(const bf16* __restrict__ A,
                                                 const bf16* __restrict__ Wm,
                                                 bf16* __restrict__ out,
                                                 int N, int K, float scale, int mode) {
  int lane = threadIdx.x & 31, wave = threadIdx.x >> 5;
  int gm0 = blockIdx.x * 128, gn0 = blockIdx.y * 128;
  v8f acc[4][4];
  zero_acc(acc);
  block_gemm_async<2, 2>(A, K, Wm, K, gm0, gn0, K, acc);
  int m0 = gm0 + (wave >> 1) * 64, n0 = gn0 + (wave & 1) * 64;
  int lhi = (lane >> 4) * 8, nn = lane & 15;
#pragma unroll
  for (int i = 0; i < 4; ++i)
#pragma unroll
    for (int j = 0; j < 4; ++j)
#pragma unroll
      for (int v = 0; v < 8; ++v) {
        int row = m0 + 16 * i + lhi + v;
        int col = n0 + 16 * j + nn;
        float val = acc[i][j][v] * scale;
        if (mode == 0) {
          out[(size_t)row * N + col] = (bf16)val;
        } else {
          int b = row >> 10, s = row & (T_DIM - 1);
          int h = col >> 6, d = col & (HEADD - 1);
          out[(((size_t)b * NH + h) * HEADD + d) * T_DIM + s] = (bf16)val;
        }
      }
}

// ---------------------------------------------------------------------------
// scores[b,h,t,s] = Q_h K_h^T (scale folded into Q), mask -> -inf, f32 out.
// K = HEAD = 64: direct loads (L2-hot), no staging.
// ---------------------------------------------------------------------------
__global__ __launch_bounds__(128) void attn_scores(const bf16* __restrict__ Q,
                                                   const bf16* __restrict__ Kb,
                                                   const unsigned char* __restrict__ mask,
                                                   float* __restrict__ out_w) {
  int z = blockIdx.z;  // b*NH + h
  int b = z >> 4, h = z & 15;
  int lane = threadIdx.x & 31, wave = threadIdx.x >> 5;
  int m0 = blockIdx.x * 128 + (wave >> 1) * 64;
  int n0 = blockIdx.y * 128 + (wave & 1) * 64;
  const bf16* Ah = Q + (size_t)b * T_DIM * EMBD + h * HEADD;
  const bf16* Bh = Kb + (size_t)b * T_DIM * EMBD + h * HEADD;
  v8f acc[4][4];
  zero_acc(acc);
  wave_gemm_64x64(Ah, EMBD, Bh, EMBD, m0, n0, HEADD, lane, acc);
  const unsigned char* mrow = mask + (size_t)b * T_DIM * T_DIM;
  float* wpage = out_w + (size_t)z * T_DIM * T_DIM;
  int lhi = (lane >> 4) * 8, nn = lane & 15;
#pragma unroll
  for (int i = 0; i < 4; ++i)
#pragma unroll
    for (int j = 0; j < 4; ++j)
#pragma unroll
      for (int v = 0; v < 8; ++v) {
        int t = m0 + 16 * i + lhi + v;
        int s = n0 + 16 * j + nn;
        float val = mrow[(size_t)t * T_DIM + s] ? -__builtin_inff() : acc[i][j][v];
        wpage[(size_t)t * T_DIM + s] = val;
      }
}

// ---------------------------------------------------------------------------
// Row softmax over 1024 cols; in-place f32 (weights output) + bf16 copy.
// Fully-masked rows -> 0 (nan_to_num semantics).
// ---------------------------------------------------------------------------
__global__ __launch_bounds__(256) void softmax_rows(float* __restrict__ w,
                                                    bf16* __restrict__ wbf) {
  size_t row = blockIdx.x;
  float* p = w + row * T_DIM;
  bf16* q = wbf + row * T_DIM;
  int t = threadIdx.x, lane = t & 31, wave = t >> 5;
  float v[4];
  float mx = -__builtin_inff();
#pragma unroll
  for (int i = 0; i < 4; ++i) { v[i] = p[t + 256 * i]; mx = fmaxf(mx, v[i]); }
  __shared__ float redm[8];
  __shared__ float reds[8];
  for (int o = 16; o > 0; o >>= 1) mx = fmaxf(mx, __shfl_xor(mx, o, 32));
  if (lane == 0) redm[wave] = mx;
  __syncthreads();
  mx = fmaxf(fmaxf(fmaxf(redm[0], redm[1]), fmaxf(redm[2], redm[3])),
             fmaxf(fmaxf(redm[4], redm[5]), fmaxf(redm[6], redm[7])));
  bool dead = !(mx > -__builtin_inff());
  float e[4];
  float sum = 0.f;
#pragma unroll
  for (int i = 0; i < 4; ++i) {
    e[i] = (dead || v[i] == -__builtin_inff()) ? 0.f : __expf(v[i] - mx);
    sum += e[i];
  }
  for (int o = 16; o > 0; o >>= 1) sum += __shfl_xor(sum, o, 32);
  if (lane == 0) reds[wave] = sum;
  __syncthreads();
  sum = reds[0] + reds[1] + reds[2] + reds[3] + reds[4] + reds[5] + reds[6] + reds[7];
  float inv = sum > 0.f ? 1.f / sum : 0.f;
#pragma unroll
  for (int i = 0; i < 4; ++i) {
    float r = e[i] * inv;
    p[t + 256 * i] = r;
    q[t + 256 * i] = (bf16)r;
  }
}

// ---------------------------------------------------------------------------
// out_h = P[T x T] * Vt[HEAD x T]^T -> merged-head bf16 [B*T, EMB].
// Block tile 256x64 (4 waves stacked in M), async-LDS staged (streams 128 MB).
// ---------------------------------------------------------------------------
__global__ __launch_bounds__(128) void attn_pv(const bf16* __restrict__ P,
                                               const bf16* __restrict__ Vt,
                                               bf16* __restrict__ attn) {
  int z = blockIdx.z;
  int b = z >> 4, h = z & 15;
  int lane = threadIdx.x & 31, wave = threadIdx.x >> 5;
  int gm0 = blockIdx.x * 256;
  const bf16* A = P + (size_t)z * T_DIM * T_DIM;
  const bf16* B = Vt + (size_t)z * HEADD * T_DIM;
  v8f acc[4][4];
  zero_acc(acc);
  block_gemm_async<4, 1>(A, T_DIM, B, T_DIM, gm0, 0, T_DIM, acc);
  int m0 = gm0 + wave * 64;
  int lhi = (lane >> 4) * 8, nn = lane & 15;
#pragma unroll
  for (int i = 0; i < 4; ++i)
#pragma unroll
    for (int j = 0; j < 4; ++j)
#pragma unroll
      for (int v = 0; v < 8; ++v) {
        int t = m0 + 16 * i + lhi + v;
        int d = 16 * j + nn;
        attn[((size_t)b * T_DIM + t) * EMBD + h * HEADD + d] = (bf16)acc[i][j][v];
      }
}

// ---------------------------------------------------------------------------
// Final projection: out_y = attn[4096 x 1024] * Wo[1024 x 1024]^T, f32 out.
// ---------------------------------------------------------------------------
__global__ __launch_bounds__(128) void gemm_out(const bf16* __restrict__ A,
                                                const bf16* __restrict__ Wm,
                                                float* __restrict__ out) {
  int lane = threadIdx.x & 31, wave = threadIdx.x >> 5;
  int gm0 = blockIdx.x * 128, gn0 = blockIdx.y * 128;
  v8f acc[4][4];
  zero_acc(acc);
  block_gemm_async<2, 2>(A, EMBD, Wm, EMBD, gm0, gn0, EMBD, acc);
  int m0 = gm0 + (wave >> 1) * 64, n0 = gn0 + (wave & 1) * 64;
  int lhi = (lane >> 4) * 8, nn = lane & 15;
#pragma unroll
  for (int i = 0; i < 4; ++i)
#pragma unroll
    for (int j = 0; j < 4; ++j)
#pragma unroll
      for (int v = 0; v < 8; ++v) {
        int row = m0 + 16 * i + lhi + v;
        int col = n0 + 16 * j + nn;
        out[(size_t)row * EMBD + col] = acc[i][j][v];
      }
}

// ---------------------------------------------------------------------------
extern "C" void kernel_launch(void* const* d_in, const int* in_sizes, int n_in,
                              void* d_out, int out_size, void* d_ws, size_t ws_size,
                              hipStream_t stream) {
  const float* x = (const float*)d_in[0];
  const float* ctx = (const float*)d_in[1];
  const unsigned char* mask = (const unsigned char*)d_in[2];  // jnp bool = 1 byte
  const float* Wq = (const float*)d_in[3];
  const float* Wk = (const float*)d_in[4];
  const float* Wv = (const float*)d_in[5];
  const float* Wo = (const float*)d_in[6];

  float* out_y = (float*)d_out;                         // [4,1024,1024]
  float* out_w = out_y + (size_t)BATCH * T_DIM * EMBD;  // [4,16,1024,1024]

  const size_t NTOK = (size_t)BATCH * T_DIM;  // 4096
  bf16* p = (bf16*)d_ws;
  bf16* x_bf = p;  p += NTOK * EMBD;
  bf16* c_bf = p;  p += NTOK * EMBD;
  bf16* wq_bf = p; p += (size_t)EMBD * EMBD;
  bf16* wk_bf = p; p += (size_t)EMBD * EMBD;
  bf16* wv_bf = p; p += (size_t)EMBD * EMBD;
  bf16* wo_bf = p; p += (size_t)EMBD * EMBD;
  bf16* q_bf = p;  p += NTOK * EMBD;
  bf16* k_bf = p;  p += NTOK * EMBD;
  bf16* vt_bf = p; p += NTOK * EMBD;  // [B,H,64,T]
  bf16* a_bf = p;  p += NTOK * EMBD;
  bf16* w_bf = p;  // [B,H,T,T] bf16 softmax copy

  int ne = (int)(NTOK * EMBD);  // 4,194,304
  int nw = EMBD * EMBD;         // 1,048,576
  dim3 cb(256);
  cvt_f32_bf16<<<dim3(2048), cb, 0, stream>>>(x, x_bf, ne);
  cvt_f32_bf16<<<dim3(2048), cb, 0, stream>>>(ctx, c_bf, ne);
  cvt_f32_bf16<<<dim3(512), cb, 0, stream>>>(Wq, wq_bf, nw);
  cvt_f32_bf16<<<dim3(512), cb, 0, stream>>>(Wk, wk_bf, nw);
  cvt_f32_bf16<<<dim3(512), cb, 0, stream>>>(Wv, wv_bf, nw);
  cvt_f32_bf16<<<dim3(512), cb, 0, stream>>>(Wo, wo_bf, nw);

  dim3 gb(128);  // 4 wave32s
  gemm_proj<<<dim3(32, 8), gb, 0, stream>>>(x_bf, wq_bf, q_bf, EMBD, EMBD, 0.125f, 0);
  gemm_proj<<<dim3(32, 8), gb, 0, stream>>>(c_bf, wk_bf, k_bf, EMBD, EMBD, 1.0f, 0);
  gemm_proj<<<dim3(32, 8), gb, 0, stream>>>(c_bf, wv_bf, vt_bf, EMBD, EMBD, 1.0f, 1);
  attn_scores<<<dim3(8, 8, BATCH * NH), gb, 0, stream>>>(q_bf, k_bf, mask, out_w);
  softmax_rows<<<dim3(BATCH * NH * T_DIM), dim3(256), 0, stream>>>(out_w, w_bf);
  attn_pv<<<dim3(4, 1, BATCH * NH), gb, 0, stream>>>(w_bf, vt_bf, a_bf);
  gemm_out<<<dim3(32, 8), gb, 0, stream>>>(a_bf, wo_bf, out_y);
}
